// DeepInterestEvolutionNetwork_38276748542532
// MI455X (gfx1250) — compile-verified, matrix operand order checked by reading, not run on hardware
//
#include <hip/hip_runtime.h>
#include <hip/hip_bf16.h>

typedef __attribute__((ext_vector_type(16))) __bf16 v16bf;
typedef __attribute__((ext_vector_type(8)))  float  v8f;

#define Bc 1024
#define Tc 200
#define Dc 128
#define Hc 128

// ---------------------------------------------------------------------------
// WMMA helpers: A row-major (M x K) bf16, W row-major (N x K) bf16.
// 16-bit A 16x32 layout: lane = half*16 + r; lane supplies row (m0+r),
// chunk0 = K[k0 + half*8 .. +7] (VGPR0-3), chunk1 = K[k0+16+half*8 .. +7].
// B 32x16 layout is symmetric with r = column index.
// ---------------------------------------------------------------------------
__device__ inline v16bf load16(const __hip_bfloat16* rowptr, int k0, int half) {
    const uint4* p = reinterpret_cast<const uint4*>(rowptr + k0 + half * 8);
    union { uint4 u[2]; v16bf v; } t;
    t.u[0] = p[0];   // 8 bf16 at k0 + half*8
    t.u[1] = p[2];   // 8 bf16 at k0 + 16 + half*8
    return t.v;
}

__device__ inline v8f wmma_bf16(v16bf a, v16bf b, v8f c) {
    return __builtin_amdgcn_wmma_f32_16x16x32_bf16(
        /*neg_a=*/false, a, /*neg_b=*/false, b,
        /*c_mod=*/(short)0, c, /*reuse_a=*/false, /*reuse_b=*/false);
}

// Software-pipelined K-loop: issue next chunk's loads before consuming the
// current chunk, so WMMAs overlap outstanding global_load_b128s (counted
// s_wait_loadcnt instead of wait-to-zero).
template <int NB>
__device__ inline void wmma_kloop(const __hip_bfloat16* arow,
                                  const __hip_bfloat16* const* brows,
                                  int K, int half, v8f* acc) {
    v16bf a_cur = load16(arow, 0, half);
    v16bf b_cur[NB];
#pragma unroll
    for (int j = 0; j < NB; ++j) b_cur[j] = load16(brows[j], 0, half);
    for (int k0 = 0; k0 < K; k0 += 32) {
        int kn = (k0 + 32 < K) ? (k0 + 32) : 0;   // last iter: harmless reload
        __builtin_prefetch(arow + kn + 32, 0, 3); // near-cache prefetch
        v16bf a_nxt = load16(arow, kn, half);
        v16bf b_nxt[NB];
#pragma unroll
        for (int j = 0; j < NB; ++j) b_nxt[j] = load16(brows[j], kn, half);
#pragma unroll
        for (int j = 0; j < NB; ++j) acc[j] = wmma_bf16(a_cur, b_cur[j], acc[j]);
        a_cur = a_nxt;
#pragma unroll
        for (int j = 0; j < NB; ++j) b_cur[j] = b_nxt[j];
    }
}

__device__ inline float sigmoidf_(float x) { return 1.0f / (1.0f + __expf(-x)); }

// ---------------------------------------------------------------------------
// Generic GEMM: C[M,N] = A[M,K] * W[N,K]^T + bias, optional relu,
// fp32 and/or bf16 output. One wave computes a 16(M) x 64(N) tile.
// ---------------------------------------------------------------------------
__global__ __launch_bounds__(256) void gemm_bias_wmma(
    const __hip_bfloat16* __restrict__ A, const __hip_bfloat16* __restrict__ W,
    const float* __restrict__ bias, float* __restrict__ Cf,
    __hip_bfloat16* __restrict__ Cb, int M, int N, int K, int relu) {
    int wave = (blockIdx.x * blockDim.x + threadIdx.x) >> 5;
    int lane = threadIdx.x & 31;
    int nTiles = N >> 6;
    if (wave >= (M >> 4) * nTiles) return;
    int m0 = (wave / nTiles) << 4;
    int nb = (wave % nTiles) << 6;
    int half = lane >> 4, r = lane & 15;

    const __hip_bfloat16* arow = A + (size_t)(m0 + r) * K;
    const __hip_bfloat16* brows[4];
#pragma unroll
    for (int j = 0; j < 4; ++j) brows[j] = W + (size_t)(nb + j * 16 + r) * K;

    v8f acc[4] = {};
    wmma_kloop<4>(arow, brows, K, half, acc);

    // C layout: VGPR i -> row m0 + i + 8*half, col nb + j*16 + r
#pragma unroll
    for (int j = 0; j < 4; ++j) {
        int n = nb + j * 16 + r;
        float bv = bias ? bias[n] : 0.0f;
#pragma unroll
        for (int i = 0; i < 8; ++i) {
            int m = m0 + i + half * 8;
            float v = acc[j][i] + bv;
            if (relu) v = fmaxf(v, 0.0f);
            if (Cf) Cf[(size_t)m * N + n] = v;
            if (Cb) Cb[(size_t)m * N + n] = __float2bfloat16(v);
        }
    }
}

// ---------------------------------------------------------------------------
// GRU step t: gh = h_prev @ Whh^T (WMMA), fused gate math.
// One wave: 16 batch rows x 16 hidden cols, with the 3 gate column blocks.
// ---------------------------------------------------------------------------
__global__ __launch_bounds__(256) void gru_step_kernel(
    const float* __restrict__ xp, const __hip_bfloat16* __restrict__ hprev_b,
    const float* __restrict__ hprev_f, const __hip_bfloat16* __restrict__ Whh,
    const float* __restrict__ bhh, float* __restrict__ hnew_f,
    __hip_bfloat16* __restrict__ hnew_b, __hip_bfloat16* __restrict__ interestB,
    int t, int first) {
    int wave = (blockIdx.x * blockDim.x + threadIdx.x) >> 5;
    int lane = threadIdx.x & 31;
    if (wave >= (Bc / 16) * (Hc / 16)) return;
    int m0 = (wave >> 3) << 4;          // batch tile
    int j0 = (wave & 7) << 4;           // hidden-col tile
    int half = lane >> 4, r = lane & 15;

    v8f acc[3] = {};
    if (!first) {
        const __hip_bfloat16* arow = hprev_b + (size_t)(m0 + r) * Hc;
        const __hip_bfloat16* brows[3];
#pragma unroll
        for (int g = 0; g < 3; ++g) brows[g] = Whh + (size_t)(g * Hc + j0 + r) * Hc;
        wmma_kloop<3>(arow, brows, Hc, half, acc);
    }
    int j = j0 + r;
    float bhr = bhh[j], bhz = bhh[Hc + j], bhn = bhh[2 * Hc + j];
#pragma unroll
    for (int i = 0; i < 8; ++i) {
        int b = m0 + i + half * 8;
        size_t xq = ((size_t)b * Tc + t) * (3 * Hc);
        float rg = sigmoidf_(xp[xq + j] + acc[0][i] + bhr);
        float zg = sigmoidf_(xp[xq + Hc + j] + acc[1][i] + bhz);
        float hn = acc[2][i] + bhn;
        float ng = tanhf(xp[xq + 2 * Hc + j] + rg * hn);
        float ho = first ? 0.0f : hprev_f[(size_t)b * Hc + j];
        float h  = (1.0f - zg) * ng + zg * ho;
        hnew_f[(size_t)b * Hc + j] = h;
        hnew_b[(size_t)b * Hc + j] = __float2bfloat16(h);
        interestB[((size_t)b * Tc + t) * Hc + j] = __float2bfloat16(h);
    }
}

// ---------------------------------------------------------------------------
// AUGRU step: attention-gated update, masked carry. Same tiling as gru_step.
// ---------------------------------------------------------------------------
__global__ __launch_bounds__(256) void augru_step_kernel(
    const float* __restrict__ xp, const __hip_bfloat16* __restrict__ hprev_b,
    const float* __restrict__ hprev_f, const __hip_bfloat16* __restrict__ Whh,
    const float* __restrict__ bhh, const float* __restrict__ att,
    const unsigned char* __restrict__ mask, float* __restrict__ hnew_f,
    __hip_bfloat16* __restrict__ hnew_b, int t, int first) {
    int wave = (blockIdx.x * blockDim.x + threadIdx.x) >> 5;
    int lane = threadIdx.x & 31;
    if (wave >= (Bc / 16) * (Hc / 16)) return;
    int m0 = (wave >> 3) << 4;
    int j0 = (wave & 7) << 4;
    int half = lane >> 4, r = lane & 15;

    v8f acc[3] = {};
    if (!first) {
        const __hip_bfloat16* arow = hprev_b + (size_t)(m0 + r) * Hc;
        const __hip_bfloat16* brows[3];
#pragma unroll
        for (int g = 0; g < 3; ++g) brows[g] = Whh + (size_t)(g * Hc + j0 + r) * Hc;
        wmma_kloop<3>(arow, brows, Hc, half, acc);
    }
    int j = j0 + r;
    float bhr = bhh[j], bhz = bhh[Hc + j], bhn = bhh[2 * Hc + j];
#pragma unroll
    for (int i = 0; i < 8; ++i) {
        int b = m0 + i + half * 8;
        size_t xq = ((size_t)b * Tc + t) * (3 * Hc);
        float rg = sigmoidf_(xp[xq + j] + acc[0][i] + bhr);
        float zg = sigmoidf_(xp[xq + Hc + j] + acc[1][i] + bhz);
        float hn = acc[2][i] + bhn;
        float ng = tanhf(xp[xq + 2 * Hc + j] + rg * hn);
        float ho = first ? 0.0f : hprev_f[(size_t)b * Hc + j];
        float a_t = att[(size_t)b * Tc + t];
        float za = a_t * zg;
        float h  = (1.0f - za) * ho + za * ng;
        if (!mask[(size_t)b * Tc + t]) h = ho;
        hnew_f[(size_t)b * Hc + j] = h;
        hnew_b[(size_t)b * Hc + j] = __float2bfloat16(h);
    }
}

// ---------------------------------------------------------------------------
// Attention stage 1: a1 = relu(feat @ W1 + b1), feat = [i, q, i-q, i*q]
// synthesized on the fly from bf16 interest / item tensors.  M=B*T, N=64, K=512.
// ---------------------------------------------------------------------------
__device__ inline uint4 feat8(const __hip_bfloat16* I, const __hip_bfloat16* Q,
                              int m, int b, int k8) {
    int region = k8 >> 7;     // 0:i 1:q 2:i-q 3:i*q
    int off = k8 & 127;
    const uint4* ip = reinterpret_cast<const uint4*>(I + (size_t)m * Dc + off);
    const uint4* qp = reinterpret_cast<const uint4*>(Q + (size_t)b * Dc + off);
    if (region == 0) return ip[0];
    if (region == 1) return qp[0];
    union U { uint4 u; __hip_bfloat16 h[8]; } iv, qv, ov;
    iv.u = ip[0]; qv.u = qp[0];
#pragma unroll
    for (int e = 0; e < 8; ++e) {
        float fi = __bfloat162float(iv.h[e]);
        float fq = __bfloat162float(qv.h[e]);
        ov.h[e] = __float2bfloat16(region == 2 ? fi - fq : fi * fq);
    }
    return ov.u;
}

__device__ inline v16bf feat16(const __hip_bfloat16* I, const __hip_bfloat16* Q,
                               int m, int b, int k0, int half) {
    union { uint4 u[2]; v16bf v; } a;
    a.u[0] = feat8(I, Q, m, b, k0 + half * 8);
    a.u[1] = feat8(I, Q, m, b, k0 + 16 + half * 8);
    return a.v;
}

__global__ __launch_bounds__(256) void att1_kernel(
    const __hip_bfloat16* __restrict__ interestB,
    const __hip_bfloat16* __restrict__ itemB,
    const __hip_bfloat16* __restrict__ W1t, const float* __restrict__ b1,
    __hip_bfloat16* __restrict__ a1) {
    int wave = (blockIdx.x * blockDim.x + threadIdx.x) >> 5;
    int lane = threadIdx.x & 31;
    if (wave >= (Bc * Tc) / 16) return;
    int m0 = wave << 4;
    int half = lane >> 4, r = lane & 15;
    int m = m0 + r;
    int b = m / Tc;

    const __hip_bfloat16* brows[4];
#pragma unroll
    for (int j = 0; j < 4; ++j) brows[j] = W1t + (size_t)(j * 16 + r) * 512;

    // pipelined: synthesize next A fragment & load next W tiles before WMMAs
    v8f acc[4] = {};
    v16bf a_cur = feat16(interestB, itemB, m, b, 0, half);
    v16bf b_cur[4];
#pragma unroll
    for (int j = 0; j < 4; ++j) b_cur[j] = load16(brows[j], 0, half);
    for (int k0 = 0; k0 < 512; k0 += 32) {
        int kn = (k0 + 32 < 512) ? (k0 + 32) : 0;
        v16bf a_nxt = feat16(interestB, itemB, m, b, kn, half);
        v16bf b_nxt[4];
#pragma unroll
        for (int j = 0; j < 4; ++j) b_nxt[j] = load16(brows[j], kn, half);
#pragma unroll
        for (int j = 0; j < 4; ++j) acc[j] = wmma_bf16(a_cur, b_cur[j], acc[j]);
        a_cur = a_nxt;
#pragma unroll
        for (int j = 0; j < 4; ++j) b_cur[j] = b_nxt[j];
    }
#pragma unroll
    for (int j = 0; j < 4; ++j) {
        int n = j * 16 + r;
        float bv = b1[n];
#pragma unroll
        for (int i = 0; i < 8; ++i) {
            int mm = m0 + i + half * 8;
            a1[(size_t)mm * 64 + n] = __float2bfloat16(fmaxf(acc[j][i] + bv, 0.0f));
        }
    }
}

// ---------------------------------------------------------------------------
// Attention stage 2: a2 = relu(a1 @ W2 + b2); scores = a2 @ Wo + bo, masked.
// Fused: per-(m,n) relu*Wo partial, then 16-lane shuffle reduction.
// ---------------------------------------------------------------------------
__global__ __launch_bounds__(256) void att2_kernel(
    const __hip_bfloat16* __restrict__ a1, const __hip_bfloat16* __restrict__ W2t,
    const float* __restrict__ b2, const float* __restrict__ Wo,
    const float* __restrict__ bo, const unsigned char* __restrict__ mask,
    float* __restrict__ scores) {
    int wave = (blockIdx.x * blockDim.x + threadIdx.x) >> 5;
    int lane = threadIdx.x & 31;
    if (wave >= (Bc * Tc) / 16) return;
    int m0 = wave << 4;
    int half = lane >> 4, r = lane & 15;

    const __hip_bfloat16* arow = a1 + (size_t)(m0 + r) * 64;
    const __hip_bfloat16* brows[2] = { W2t + (size_t)r * 64,
                                       W2t + (size_t)(16 + r) * 64 };
    v8f acc[2] = {};
    wmma_kloop<2>(arow, brows, 64, half, acc);

    float wo0 = Wo[r], wo1 = Wo[16 + r];
    float bb0 = b2[r], bb1 = b2[16 + r];
#pragma unroll
    for (int i = 0; i < 8; ++i) {
        float p = fmaxf(acc[0][i] + bb0, 0.0f) * wo0 + fmaxf(acc[1][i] + bb1, 0.0f) * wo1;
#pragma unroll
        for (int s = 1; s < 16; s <<= 1) p += __shfl_xor(p, s, 32);
        if (r == 0) {
            int m = m0 + i + half * 8;
            scores[m] = mask[m] ? (p + bo[0]) : -1e9f;
        }
    }
}

// Row softmax over T=200, one wave per batch row.
__global__ __launch_bounds__(256) void softmax_kernel(const float* __restrict__ scores,
                                                      float* __restrict__ att) {
    int b = (blockIdx.x * blockDim.x + threadIdx.x) >> 5;
    int lane = threadIdx.x & 31;
    if (b >= Bc) return;
    const float* s = scores + (size_t)b * Tc;
    float mx = -1e30f;
    for (int t = lane; t < Tc; t += 32) mx = fmaxf(mx, s[t]);
    for (int d = 1; d < 32; d <<= 1) mx = fmaxf(mx, __shfl_xor(mx, d, 32));
    float sum = 0.0f;
    for (int t = lane; t < Tc; t += 32) sum += __expf(s[t] - mx);
    for (int d = 1; d < 32; d <<= 1) sum += __shfl_xor(sum, d, 32);
    float inv = 1.0f / sum;
    float* o = att + (size_t)b * Tc;
    for (int t = lane; t < Tc; t += 32) o[t] = __expf(s[t] - mx) * inv;
}

// ---------------------------------------------------------------------------
// Gathers / converts / pack / output
// ---------------------------------------------------------------------------
__global__ void gather_hist_kernel(const int* __restrict__ items,
                                   const float* __restrict__ emb,
                                   __hip_bfloat16* __restrict__ histB) {
    size_t idx = (size_t)blockIdx.x * blockDim.x + threadIdx.x;
    if (idx >= (size_t)Bc * Tc * (Dc / 8)) return;
    size_t m = idx >> 4;
    int c = (int)(idx & 15) * 8;
    const float* e = emb + (size_t)items[m] * Dc + c;
    union { uint4 u; __hip_bfloat16 h[8]; } o;
#pragma unroll
    for (int i = 0; i < 8; ++i) o.h[i] = __float2bfloat16(e[i]);
    *reinterpret_cast<uint4*>(histB + m * Dc + c) = o.u;
}

__global__ void gather_item_kernel(const int* __restrict__ items,
                                   const float* __restrict__ emb,
                                   float* __restrict__ itemF,
                                   __hip_bfloat16* __restrict__ itemB) {
    int idx = blockIdx.x * blockDim.x + threadIdx.x;
    if (idx >= Bc * (Dc / 8)) return;
    int b = idx >> 4, c = (idx & 15) * 8;
    const float* e = emb + (size_t)items[b] * Dc + c;
#pragma unroll
    for (int i = 0; i < 8; ++i) {
        float v = e[i];
        itemF[(size_t)b * Dc + c + i] = v;
        itemB[(size_t)b * Dc + c + i] = __float2bfloat16(v);
    }
}

__global__ void cvt_bf16_kernel(const float* __restrict__ src,
                                __hip_bfloat16* __restrict__ dst, int n) {
    int i = blockIdx.x * blockDim.x + threadIdx.x;
    if (i < n) dst[i] = __float2bfloat16(src[i]);
}

// src (R,C) row-major -> dst (C,R) row-major, bf16
__global__ void cvt_t_bf16_kernel(const float* __restrict__ src,
                                  __hip_bfloat16* __restrict__ dst, int R, int C) {
    int i = blockIdx.x * blockDim.x + threadIdx.x;
    if (i >= R * C) return;
    int c = i / R, r = i % R;
    dst[(size_t)c * R + r] = __float2bfloat16(src[(size_t)r * C + c]);
}

__global__ void pack_mlpin_kernel(const float* __restrict__ hfin,
                                  const __hip_bfloat16* __restrict__ itemB,
                                  __hip_bfloat16* __restrict__ mlpin) {
    int idx = blockIdx.x * blockDim.x + threadIdx.x;
    if (idx >= Bc * 256) return;
    int b = idx >> 8, k = idx & 255;
    mlpin[idx] = (k < Hc) ? __float2bfloat16(hfin[(size_t)b * Hc + k])
                          : itemB[(size_t)b * Dc + (k - Hc)];
}

__global__ __launch_bounds__(256) void out_kernel(const float* __restrict__ h2,
                                                  const float* __restrict__ outW,
                                                  const float* __restrict__ outB,
                                                  float* __restrict__ out) {
    int b = (blockIdx.x * blockDim.x + threadIdx.x) >> 5;
    int lane = threadIdx.x & 31;
    if (b >= Bc) return;
    const float* row = h2 + (size_t)b * 128;
    float s = 0.0f;
#pragma unroll
    for (int q = 0; q < 4; ++q) s += row[q * 32 + lane] * outW[q * 32 + lane];
    for (int d = 1; d < 32; d <<= 1) s += __shfl_xor(s, d, 32);
    if (lane == 0) out[b] = s + outB[0];
}

// ---------------------------------------------------------------------------
// Host orchestration
// ---------------------------------------------------------------------------
extern "C" void kernel_launch(void* const* d_in, const int* in_sizes, int n_in,
                              void* d_out, int out_size, void* d_ws, size_t ws_size,
                              hipStream_t stream) {
    const int*   target  = (const int*)d_in[0];
    const int*   hist    = (const int*)d_in[1];
    const unsigned char* mask = (const unsigned char*)d_in[2];
    const float* emb     = (const float*)d_in[3];
    const float* gWih    = (const float*)d_in[4];
    const float* gWhh    = (const float*)d_in[5];
    const float* gbih    = (const float*)d_in[6];
    const float* gbhh    = (const float*)d_in[7];
    const float* aWih    = (const float*)d_in[8];
    const float* aWhh    = (const float*)d_in[9];
    const float* abih    = (const float*)d_in[10];
    const float* abhh    = (const float*)d_in[11];
    const float* attW1   = (const float*)d_in[12];
    const float* attb1   = (const float*)d_in[13];
    const float* attW2   = (const float*)d_in[14];
    const float* attb2   = (const float*)d_in[15];
    const float* attWo   = (const float*)d_in[16];
    const float* attbo   = (const float*)d_in[17];
    const float* mW1     = (const float*)d_in[18];
    const float* mb1     = (const float*)d_in[19];
    const float* mW2     = (const float*)d_in[20];
    const float* mb2     = (const float*)d_in[21];
    const float* outW    = (const float*)d_in[22];
    const float* outB    = (const float*)d_in[23];
    float* out = (float*)d_out;

    char* base = (char*)d_ws;
    size_t off = 0;
    auto alloc = [&](size_t bytes) -> char* {
        char* p = base + off;
        off = (off + bytes + 255) & ~(size_t)255;
        return p;
    };
    const size_t MT = (size_t)Bc * Tc;
    __hip_bfloat16* histB     = (__hip_bfloat16*)alloc(MT * Dc * 2);
    __hip_bfloat16* interestB = (__hip_bfloat16*)alloc(MT * Hc * 2);
    float*          itemF     = (float*)alloc((size_t)Bc * Dc * 4);
    __hip_bfloat16* itemB     = (__hip_bfloat16*)alloc((size_t)Bc * Dc * 2);
    float*          xp        = (float*)alloc(MT * 3 * Hc * 4);
    float*          hAf       = (float*)alloc((size_t)Bc * Hc * 4);
    float*          hBf       = (float*)alloc((size_t)Bc * Hc * 4);
    __hip_bfloat16* hAb       = (__hip_bfloat16*)alloc((size_t)Bc * Hc * 2);
    __hip_bfloat16* hBb       = (__hip_bfloat16*)alloc((size_t)Bc * Hc * 2);
    __hip_bfloat16* a1        = (__hip_bfloat16*)alloc(MT * 64 * 2);
    float*          scores    = (float*)alloc(MT * 4);
    float*          attP      = (float*)alloc(MT * 4);
    __hip_bfloat16* mlpin     = (__hip_bfloat16*)alloc((size_t)Bc * 256 * 2);
    __hip_bfloat16* h1        = (__hip_bfloat16*)alloc((size_t)Bc * 256 * 2);
    float*          h2        = (float*)alloc((size_t)Bc * 128 * 4);
    __hip_bfloat16* wGih = (__hip_bfloat16*)alloc(384 * 128 * 2);
    __hip_bfloat16* wGhh = (__hip_bfloat16*)alloc(384 * 128 * 2);
    __hip_bfloat16* wAih = (__hip_bfloat16*)alloc(384 * 128 * 2);
    __hip_bfloat16* wAhh = (__hip_bfloat16*)alloc(384 * 128 * 2);
    __hip_bfloat16* wA1t = (__hip_bfloat16*)alloc(64 * 512 * 2);
    __hip_bfloat16* wA2t = (__hip_bfloat16*)alloc(32 * 64 * 2);
    __hip_bfloat16* wM1t = (__hip_bfloat16*)alloc(256 * 256 * 2);
    __hip_bfloat16* wM2t = (__hip_bfloat16*)alloc(128 * 256 * 2);

    auto cdiv = [](long long a, long long b) { return (int)((a + b - 1) / b); };

    // Weight conversions (N x K layouts for WMMA B operand)
    cvt_bf16_kernel<<<cdiv(384 * 128, 256), 256, 0, stream>>>(gWih, wGih, 384 * 128);
    cvt_bf16_kernel<<<cdiv(384 * 128, 256), 256, 0, stream>>>(gWhh, wGhh, 384 * 128);
    cvt_bf16_kernel<<<cdiv(384 * 128, 256), 256, 0, stream>>>(aWih, wAih, 384 * 128);
    cvt_bf16_kernel<<<cdiv(384 * 128, 256), 256, 0, stream>>>(aWhh, wAhh, 384 * 128);
    cvt_t_bf16_kernel<<<cdiv(512 * 64, 256), 256, 0, stream>>>(attW1, wA1t, 512, 64);
    cvt_t_bf16_kernel<<<cdiv(64 * 32, 256), 256, 0, stream>>>(attW2, wA2t, 64, 32);
    cvt_t_bf16_kernel<<<cdiv(256 * 256, 256), 256, 0, stream>>>(mW1, wM1t, 256, 256);
    cvt_t_bf16_kernel<<<cdiv(256 * 128, 256), 256, 0, stream>>>(mW2, wM2t, 256, 128);

    // Embedding gathers (fp32 -> bf16)
    gather_hist_kernel<<<cdiv(MT * 16, 256), 256, 0, stream>>>(hist, emb, histB);
    gather_item_kernel<<<cdiv(Bc * 16, 256), 256, 0, stream>>>(target, emb, itemF, itemB);

    auto launch_gemm = [&](const __hip_bfloat16* A, const __hip_bfloat16* W,
                           const float* bias, float* Cf, __hip_bfloat16* Cb,
                           int M, int N, int K, int relu) {
        long long waves = (long long)(M / 16) * (N / 64);
        gemm_bias_wmma<<<cdiv(waves * 32, 256), 256, 0, stream>>>(A, W, bias, Cf, Cb,
                                                                  M, N, K, relu);
    };

    // GRU input projection: xp = hist_emb @ Wih^T + bih   (204800 x 384, K=128)
    launch_gemm(histB, wGih, gbih, xp, nullptr, Bc * Tc, 3 * Hc, Dc, 0);

    // GRU scan (200 sequential step kernels; double-buffered h)
    for (int t = 0; t < Tc; ++t) {
        const float* hin_f = (t & 1) ? hBf : hAf;
        const __hip_bfloat16* hin_b = (t & 1) ? hBb : hAb;
        float* hout_f = (t & 1) ? hAf : hBf;
        __hip_bfloat16* hout_b = (t & 1) ? hAb : hBb;
        gru_step_kernel<<<64, 256, 0, stream>>>(xp, hin_b, hin_f, wGhh, gbhh,
                                                hout_f, hout_b, interestB, t, t == 0);
    }

    // Attention
    att1_kernel<<<cdiv((long long)(MT / 16) * 32, 256), 256, 0, stream>>>(
        interestB, itemB, wA1t, attb1, a1);
    att2_kernel<<<cdiv((long long)(MT / 16) * 32, 256), 256, 0, stream>>>(
        a1, wA2t, attb2, attWo, attbo, mask, scores);
    softmax_kernel<<<cdiv(Bc * 32, 256), 256, 0, stream>>>(scores, attP);

    // AUGRU input projection (reuses xp buffer): xp = interest @ augru_Wih^T + bih
    launch_gemm(interestB, wAih, abih, xp, nullptr, Bc * Tc, 3 * Hc, Hc, 0);

    // AUGRU scan
    for (int t = 0; t < Tc; ++t) {
        const float* hin_f = (t & 1) ? hBf : hAf;
        const __hip_bfloat16* hin_b = (t & 1) ? hBb : hAb;
        float* hout_f = (t & 1) ? hAf : hBf;
        __hip_bfloat16* hout_b = (t & 1) ? hAb : hBb;
        augru_step_kernel<<<64, 256, 0, stream>>>(xp, hin_b, hin_f, wAhh, abhh,
                                                  attP, mask, hout_f, hout_b, t, t == 0);
    }
    // t=199 (odd) wrote hAf -> final hidden state lives in hAf

    // Final MLP
    pack_mlpin_kernel<<<cdiv(Bc * 256, 256), 256, 0, stream>>>(hAf, itemB, mlpin);
    launch_gemm(mlpin, wM1t, mb1, nullptr, h1, Bc, 256, 256, 1);
    launch_gemm(h1, wM2t, mb2, h2, nullptr, Bc, 128, 256, 1);
    out_kernel<<<cdiv(Bc * 32, 256), 256, 0, stream>>>(h2, outW, outB, out);

    (void)in_sizes; (void)n_in; (void)out_size; (void)ws_size;
}
